// W8Linear_18502719111670
// MI455X (gfx1250) — matrix operand order repaired and checked by reading, not compiled, not run on hardware
//
#include <hip/hip_runtime.h>

// Problem constants (from reference): B=4, S=2048, D_IN=4096, D_OUT=11008
#define M_DIM 8192      // B*S
#define N_DIM 11008     // D_OUT
#define K_DIM 4096      // D_IN

typedef __attribute__((ext_vector_type(16))) __bf16      v16bf;
typedef __attribute__((ext_vector_type(8)))  __bf16      v8bf;
typedef __attribute__((ext_vector_type(8)))  float       v8f;
typedef __attribute__((ext_vector_type(4)))  unsigned int v4u;

// ---------------------------------------------------------------------------
// Prep kernel 1: int32 weight (values in [-127,127]) -> bf16 (exact).
// 8 elements per thread, b128-in / b128-out.
// ---------------------------------------------------------------------------
__global__ __launch_bounds__(256) void w8_cvt_weight(const int* __restrict__ wq,
                                                     __bf16* __restrict__ wbf,
                                                     unsigned int n8) {
    unsigned int t = blockIdx.x * blockDim.x + threadIdx.x;
    if (t >= n8) return;
    size_t idx = (size_t)t * 8;
    v4u a = *(const v4u*)(wq + idx);
    v4u b = *(const v4u*)(wq + idx + 4);
    v8bf o;
    o[0] = (__bf16)(float)(int)a.x;  o[1] = (__bf16)(float)(int)a.y;
    o[2] = (__bf16)(float)(int)a.z;  o[3] = (__bf16)(float)(int)a.w;
    o[4] = (__bf16)(float)(int)b.x;  o[5] = (__bf16)(float)(int)b.y;
    o[6] = (__bf16)(float)(int)b.z;  o[7] = (__bf16)(float)(int)b.w;
    *(v8bf*)(wbf + idx) = o;
}

// ---------------------------------------------------------------------------
// Prep kernel 2: f32 activation -> bf16.
// ---------------------------------------------------------------------------
__global__ __launch_bounds__(256) void w8_cvt_x(const float* __restrict__ x,
                                                __bf16* __restrict__ xbf,
                                                unsigned int n8) {
    unsigned int t = blockIdx.x * blockDim.x + threadIdx.x;
    if (t >= n8) return;
    size_t idx = (size_t)t * 8;
    v4u a = *(const v4u*)(x + idx);
    v4u b = *(const v4u*)(x + idx + 4);
    const float* fa = (const float*)&a;
    const float* fb = (const float*)&b;
    v8bf o;
    o[0] = (__bf16)fa[0]; o[1] = (__bf16)fa[1]; o[2] = (__bf16)fa[2]; o[3] = (__bf16)fa[3];
    o[4] = (__bf16)fb[0]; o[5] = (__bf16)fb[1]; o[6] = (__bf16)fb[2]; o[7] = (__bf16)fb[3];
    *(v8bf*)(xbf + idx) = o;
}

// ---------------------------------------------------------------------------
// Fragment loaders matching the CDNA5 WMMA VGPR layouts (CDNA5 ISA).
//
// A (16x32 bf16): lane l holds row M = l%16; K chunks are contiguous 8-element
// runs at K = (l<16?0:8) and +16.  -> two b128 loads, 16 elements apart.
// B (32x16 bf16): lane l holds column N = l%16; K chunks are contiguous 8-element
// runs at K = (l<16?0:8) and +16.  -> two b128 loads, 16 elements apart.
// ---------------------------------------------------------------------------
__device__ __forceinline__ v16bf load_frag_a(const __bf16* p) {
    union { v4u q[2]; v16bf v; } u;
    u.q[0] = *(const v4u*)(p);        // K kk .. kk+7
    u.q[1] = *(const v4u*)(p + 16);   // K kk+16 .. kk+23
    return u.v;
}
__device__ __forceinline__ v16bf load_frag_b(const __bf16* p) {
    union { v4u q[2]; v16bf v; } u;
    u.q[0] = *(const v4u*)(p);        // K koff .. koff+7
    u.q[1] = *(const v4u*)(p + 16);   // K koff+16 .. koff+23
    return u.v;
}

// ---------------------------------------------------------------------------
// GEMM: out[m][n] = (sum_k xbf[m][k]*wbf[n][k]) * max_val[n]/127
// Block = 128 threads = 4 waves; block tile 64(M) x 128(N);
// wave tile 32(M) x 64(N) = 2x4 grid of 16x16x32 WMMA accumulators.
// Grid: (8192/64, 11008/128) = (128, 86).
// ---------------------------------------------------------------------------
__global__ __launch_bounds__(128) void w8_gemm_wmma(
        const __bf16* __restrict__ xbf,
        const __bf16* __restrict__ wbf,
        const float*  __restrict__ max_val,
        float* __restrict__ out) {
    const int lane  = threadIdx.x & 31;
    const int wid   = threadIdx.x >> 5;
    const int wr    = wid >> 1;          // wave row (0..1)
    const int wc    = wid & 1;           // wave col (0..1)
    const int lhalf = lane & 15;
    const int hi    = lane >> 4;         // 0: lanes 0-15, 1: lanes 16-31

    const int m_base = blockIdx.x * 64 + wr * 32;
    const int n_base = blockIdx.y * 128 + wc * 64;

    // Per-lane base pointers for the K=0 fragments.
    const __bf16* arow[2];
    arow[0] = xbf + (size_t)(m_base + lhalf) * K_DIM + hi * 8;
    arow[1] = arow[0] + (size_t)16 * K_DIM;

    const __bf16* brow[4];
    brow[0] = wbf + (size_t)(n_base + lhalf) * K_DIM + hi * 8;
    brow[1] = brow[0] + (size_t)16 * K_DIM;
    brow[2] = brow[0] + (size_t)32 * K_DIM;
    brow[3] = brow[0] + (size_t)48 * K_DIM;

    v8f acc[2][4] = {};

    #pragma unroll 2
    for (int k0 = 0; k0 < K_DIM; k0 += 32) {
        v16bf a0 = load_frag_a(arow[0] + k0);
        v16bf a1 = load_frag_a(arow[1] + k0);
        v16bf b0 = load_frag_b(brow[0] + k0);
        v16bf b1 = load_frag_b(brow[1] + k0);
        v16bf b2 = load_frag_b(brow[2] + k0);
        v16bf b3 = load_frag_b(brow[3] + k0);

        acc[0][0] = __builtin_amdgcn_wmma_f32_16x16x32_bf16(false, a0, false, b0, (short)0, acc[0][0], false, false);
        acc[0][1] = __builtin_amdgcn_wmma_f32_16x16x32_bf16(false, a0, false, b1, (short)0, acc[0][1], false, false);
        acc[0][2] = __builtin_amdgcn_wmma_f32_16x16x32_bf16(false, a0, false, b2, (short)0, acc[0][2], false, false);
        acc[0][3] = __builtin_amdgcn_wmma_f32_16x16x32_bf16(false, a0, false, b3, (short)0, acc[0][3], false, false);
        acc[1][0] = __builtin_amdgcn_wmma_f32_16x16x32_bf16(false, a1, false, b0, (short)0, acc[1][0], false, false);
        acc[1][1] = __builtin_amdgcn_wmma_f32_16x16x32_bf16(false, a1, false, b1, (short)0, acc[1][1], false, false);
        acc[1][2] = __builtin_amdgcn_wmma_f32_16x16x32_bf16(false, a1, false, b2, (short)0, acc[1][2], false, false);
        acc[1][3] = __builtin_amdgcn_wmma_f32_16x16x32_bf16(false, a1, false, b3, (short)0, acc[1][3], false, false);
    }

    // Epilogue: C/D layout — VGPR r, lane l: M = r + (l>=16 ? 8:0), N = l%16.
    #pragma unroll
    for (int j = 0; j < 4; ++j) {
        const int n = n_base + j * 16 + lhalf;
        const float s = max_val[n] * (1.0f / 127.0f);
        #pragma unroll
        for (int i = 0; i < 2; ++i) {
            const int mrow = m_base + i * 16 + hi * 8;
            float* op = out + (size_t)mrow * N_DIM + n;
            #pragma unroll
            for (int r = 0; r < 8; ++r) {
                op[(size_t)r * N_DIM] = acc[i][j][r] * s;
            }
        }
    }
}

// ---------------------------------------------------------------------------
// Host-side launch. Workspace layout (needs ~150 MiB of d_ws):
//   [0, 86 MiB)        : bf16 weights  (N_DIM*K_DIM)
//   [86 MiB, 150 MiB)  : bf16 activations (M_DIM*K_DIM)
// ---------------------------------------------------------------------------
extern "C" void kernel_launch(void* const* d_in, const int* in_sizes, int n_in,
                              void* d_out, int out_size, void* d_ws, size_t ws_size,
                              hipStream_t stream) {
    const float* x   = (const float*)d_in[0];
    const int*   wq  = (const int*)d_in[1];
    const float* mv  = (const float*)d_in[2];
    float*       out = (float*)d_out;

    __bf16* wbf = (__bf16*)d_ws;
    __bf16* xbf = (__bf16*)((char*)d_ws + (size_t)N_DIM * K_DIM * sizeof(__bf16));

    // Weight conversion: N*K / 8 threads.
    {
        unsigned int n8 = (unsigned int)(((size_t)N_DIM * K_DIM) / 8);
        w8_cvt_weight<<<(n8 + 255) / 256, 256, 0, stream>>>(wq, wbf, n8);
    }
    // Activation conversion: M*K / 8 threads.
    {
        unsigned int n8 = (unsigned int)(((size_t)M_DIM * K_DIM) / 8);
        w8_cvt_x<<<(n8 + 255) / 256, 256, 0, stream>>>(x, xbf, n8);
    }
    // WMMA GEMM.
    dim3 grid(M_DIM / 64, N_DIM / 128);
    w8_gemm_wmma<<<grid, 128, 0, stream>>>(xbf, wbf, mv, out);
}
